// GD_SSM_70325794504926
// MI455X (gfx1250) — compile-verified, matrix operand order checked by reading, not run on hardware
//
#include <hip/hip_runtime.h>
#include <hip/hip_bf16.h>

typedef _Float16 v8h  __attribute__((ext_vector_type(8)));
typedef _Float16 v16h __attribute__((ext_vector_type(16)));
typedef float    v8f  __attribute__((ext_vector_type(8)));

#define L_SEQ   8192
#define H_DIM   128
#define WIN     3
#define T_WIN   4095          // (8192-3)/2+1
#define CHUNK   8             // power of two (cross-scan uses repeated squaring)
#define NCH     512           // ceil(T_WIN/CHUNK)
#define HH      (H_DIM*H_DIM) // 16384
#define LDSP    136           // padded LDS row stride in halfs (272B, 16B-aligned)

// ---------------- Kernel 0: C_tilde f32 -> f16 copy in ws ----------------
__global__ void k_convert_c(const float* __restrict__ C, _Float16* __restrict__ Ch) {
    int i = blockIdx.x * 256 + threadIdx.x;
    if (i < HH) Ch[i] = (_Float16)C[i];
}

// ------------- Kernel 1: per-chunk local scan -> chunk finals; os -------------
__global__ __launch_bounds__(256)
void k_chunk_final(const float* __restrict__ X, const float* __restrict__ Lam,
                   const float* __restrict__ wq, const float* __restrict__ Dv,
                   float* __restrict__ wsCarry, float* __restrict__ osOut) {
    __shared__ float xw[WIN][H_DIM];
    __shared__ float g [WIN][H_DIM];
    __shared__ float wqs[WIN * WIN];
    __shared__ float Ds[WIN];
    const int tid = threadIdx.x;
    const int c   = blockIdx.x;
    if (tid < WIN * WIN) wqs[tid] = wq[tid];
    if (tid < WIN)       Ds[tid]  = Dv[tid];
    __syncthreads();

    const int d     = tid & (H_DIM - 1);
    const int abase = tid >> 7;           // 0 or 1
    float carry[64];
#pragma unroll
    for (int j = 0; j < 64; ++j) carry[j] = 0.0f;

    for (int k = 0; k < CHUNK; ++k) {
        const int t = c * CHUNK + k;
        if (t >= T_WIN) break;            // uniform per block
        if (tid < H_DIM) {
            float x0 = X[(2 * t + 0) * H_DIM + tid];
            float x1 = X[(2 * t + 1) * H_DIM + tid];
            float x2 = X[(2 * t + 2) * H_DIM + tid];
            xw[0][tid] = x0; xw[1][tid] = x1; xw[2][tid] = x2;
            g[0][tid] = wqs[0] * x0 + wqs[1] * x1 + wqs[2] * x2;
            g[1][tid] = wqs[3] * x0 + wqs[4] * x1 + wqs[5] * x2;
            g[2][tid] = wqs[6] * x0 + wqs[7] * x1 + wqs[8] * x2;
            osOut[t * H_DIM + tid] = Ds[0] * x0 + Ds[1] * x1 + Ds[2] * x2;
        }
        __syncthreads();
        const float g0 = g[0][d], g1 = g[1][d], g2 = g[2][d];
#pragma unroll
        for (int j = 0; j < 64; ++j) {
            const int a = abase + 2 * j;
            float lsa = xw[0][a] * g0 + xw[1][a] * g1 + xw[2][a] * g2;
            carry[j]  = Lam[a * H_DIM + d] * carry[j] + lsa;
        }
        __syncthreads();
    }
#pragma unroll
    for (int j = 0; j < 64; ++j)
        wsCarry[(size_t)c * HH + (abase + 2 * j) * H_DIM + d] = carry[j];
}

// ------ Kernel 2: cross-chunk scan, in place: wsCarry[c] becomes carryIn[c] ------
__global__ void k_cross_scan(const float* __restrict__ Lam, float* __restrict__ wsCarry) {
    const int e = blockIdx.x * 256 + threadIdx.x;
    if (e >= HH) return;
    float lS = Lam[e];
#pragma unroll
    for (int s = 1; s < CHUNK; s <<= 1) lS = lS * lS;   // Lambda^CHUNK exactly
    float carry = 0.0f;
    for (int c = 0; c < NCH; ++c) {
        float fin = wsCarry[(size_t)c * HH + e];
        wsCarry[(size_t)c * HH + e] = carry;   // incoming carry for chunk c
        carry = lS * carry + fin;
    }
}

// -------- Kernel 3: seeded scan + per-t GEMM  out[t] = C(h,d) x zs[t]^T(d,a) --------
__global__ __launch_bounds__(256)
void k_main(const float* __restrict__ X, const float* __restrict__ Lam,
            const float* __restrict__ wq, const _Float16* __restrict__ Ch,
            const float* __restrict__ wsCarry, float* __restrict__ Out) {
    __shared__ __align__(16) _Float16 Zl[H_DIM * LDSP];   // zs[t] as f16, row a, col d
    __shared__ float xw[WIN][H_DIM];
    __shared__ float g [WIN][H_DIM];
    __shared__ float wqs[WIN * WIN];
    const int tid = threadIdx.x;
    const int c   = blockIdx.x;
    if (tid < WIN * WIN) wqs[tid] = wq[tid];
    __syncthreads();

    const int d     = tid & (H_DIM - 1);
    const int abase = tid >> 7;
    float carry[64];
#pragma unroll
    for (int j = 0; j < 64; ++j)                       // seed with true incoming carry
        carry[j] = wsCarry[(size_t)c * HH + (abase + 2 * j) * H_DIM + d];

    const int lane = tid & 31;
    const int wave = tid >> 5;
    const int h0   = wave * 16;          // each wave owns one 16-row output stripe
    const int m    = lane & 15;          // fragment row/col index
    const int kb   = (lane >> 4) * 8;    // K base per half-wave (ISA 16-bit A/B layout)
    const int hsel = (lane >> 4) * 8;    // D-matrix M offset for upper half-wave

    for (int k = 0; k < CHUNK; ++k) {
        const int t = c * CHUNK + k;
        if (t >= T_WIN) break;
        // ---- phase 1: window rows + g = w_q @ x ----
        if (tid < H_DIM) {
            float x0 = X[(2 * t + 0) * H_DIM + tid];
            float x1 = X[(2 * t + 1) * H_DIM + tid];
            float x2 = X[(2 * t + 2) * H_DIM + tid];
            xw[0][tid] = x0; xw[1][tid] = x1; xw[2][tid] = x2;
            g[0][tid] = wqs[0] * x0 + wqs[1] * x1 + wqs[2] * x2;
            g[1][tid] = wqs[3] * x0 + wqs[4] * x1 + wqs[5] * x2;
            g[2][tid] = wqs[6] * x0 + wqs[7] * x1 + wqs[8] * x2;
        }
        __syncthreads();
        // ---- phase 2: lsa, scan update, zs -> LDS (f16) ----
        const float g0 = g[0][d], g1 = g[1][d], g2 = g[2][d];
#pragma unroll
        for (int j = 0; j < 64; ++j) {
            const int a = abase + 2 * j;
            float lsa = xw[0][a] * g0 + xw[1][a] * g1 + xw[2][a] * g2;
            float cv  = Lam[a * H_DIM + d] * carry[j] + lsa;
            carry[j]  = cv;
            Zl[a * LDSP + d] = (_Float16)cv;
        }
        __syncthreads();
        // ---- prefetch the two new window rows for t+1 into WGP$/L2 ----
        if (tid < H_DIM) {
            const int r = 2 * t + 3;
            if (r + 1 < L_SEQ) {
                __builtin_prefetch(&X[(size_t)r * H_DIM + tid], 0, 3);
                __builtin_prefetch(&X[(size_t)(r + 1) * H_DIM + tid], 0, 3);
            }
        }
        // ---- phase 3: WMMA GEMM in two groups of 4 tiles (lower acc pressure) ----
        float* outT = Out + (size_t)t * HH;
#pragma unroll
        for (int grp = 0; grp < 2; ++grp) {
            const int aBase = grp * 64;                  // tiles a0 = aBase .. aBase+48
            v8f acc[4];
#pragma unroll
            for (int ti = 0; ti < 4; ++ti)
                acc[ti] = (v8f){0.f, 0.f, 0.f, 0.f, 0.f, 0.f, 0.f, 0.f};

#pragma unroll
            for (int d0 = 0; d0 < H_DIM; d0 += 32) {
                // A fragment: C_f16 rows h0..h0+15, K-slice d0..d0+31 (L2-resident)
                const v8h* ap = (const v8h*)(Ch + (h0 + m) * H_DIM + d0 + kb);
                v8h aLo = ap[0];
                v8h aHi = ap[2];   // +16 halfs
                v16h afrag = __builtin_shufflevector(aLo, aHi, 0, 1, 2, 3, 4, 5, 6, 7,
                                                     8, 9, 10, 11, 12, 13, 14, 15);
                // double-buffered B fragments from LDS
                const v8h* bp0 = (const v8h*)(&Zl[(aBase + m) * LDSP + d0 + kb]);
                v8h bLo = bp0[0];
                v8h bHi = bp0[2];
#pragma unroll
                for (int ti = 0; ti < 4; ++ti) {
                    v8h nLo, nHi;
                    if (ti < 3) {                       // preload next B before WMMA
                        const v8h* bpn =
                            (const v8h*)(&Zl[(aBase + (ti + 1) * 16 + m) * LDSP + d0 + kb]);
                        nLo = bpn[0];
                        nHi = bpn[2];
                    }
                    v16h bfrag = __builtin_shufflevector(bLo, bHi, 0, 1, 2, 3, 4, 5, 6, 7,
                                                         8, 9, 10, 11, 12, 13, 14, 15);
                    acc[ti] = __builtin_amdgcn_wmma_f32_16x16x32_f16(
                        false, afrag, false, bfrag, (short)0, acc[ti], false, false);
                    if (ti < 3) { bLo = nLo; bHi = nHi; }
                }
            }
            // ---- store this group's D tiles (overlaps with next group's WMMAs) ----
#pragma unroll
            for (int ti = 0; ti < 4; ++ti) {
                const int a = aBase + ti * 16 + (lane & 15);
#pragma unroll
                for (int vr = 0; vr < 8; ++vr)
                    outT[(h0 + vr + hsel) * H_DIM + a] = acc[ti][vr];
            }
        }
        __syncthreads();
    }
}

extern "C" void kernel_launch(void* const* d_in, const int* in_sizes, int n_in,
                              void* d_out, int out_size, void* d_ws, size_t ws_size,
                              hipStream_t stream) {
    (void)in_sizes; (void)n_in; (void)out_size; (void)ws_size;
    const float* X   = (const float*)d_in[0];   // (L, H)
    const float* Lam = (const float*)d_in[1];   // (H, H)
    const float* C   = (const float*)d_in[2];   // (H, H)
    const float* wqp = (const float*)d_in[3];   // (3, 3)
    const float* Dv  = (const float*)d_in[4];   // (3,)

    float* Out   = (float*)d_out;                       // zs_out: T*H*H
    float* osOut = Out + (size_t)T_WIN * HH;            // os:     T*H

    float*     wsCarry = (float*)d_ws;                                    // NCH*HH f32 = 32 MB
    _Float16*  Ch      = (_Float16*)((char*)d_ws + (size_t)NCH * HH * 4); // 32 KB f16 C

    k_convert_c <<<(HH + 255) / 256, 256, 0, stream>>>(C, Ch);
    k_chunk_final<<<NCH, 256, 0, stream>>>(X, Lam, wqp, Dv, wsCarry, osOut);
    k_cross_scan <<<(HH + 255) / 256, 256, 0, stream>>>(Lam, wsCarry);
    k_main       <<<NCH, 256, 0, stream>>>(X, Lam, wqp, Ch, wsCarry, Out);
}
